// GeneralSelfAttention_10230612099232
// MI455X (gfx1250) — matrix operand order repaired and checked by reading, not compiled, not run on hardware
//
#include <hip/hip_runtime.h>
#include <hip/hip_bf16.h>
#include <cstddef>

typedef __bf16 bf16;
typedef __attribute__((ext_vector_type(16))) __bf16 v16bf;
typedef __attribute__((ext_vector_type(8)))  __bf16 v8bf;
typedef __attribute__((ext_vector_type(8)))  float  v8f;
typedef __attribute__((ext_vector_type(4)))  unsigned int uint32x4;
typedef __attribute__((ext_vector_type(8)))  int          int32x8;
typedef __attribute__((ext_vector_type(4)))  int          int32x4;

union Frag { v16bf v; v8bf h[2]; };

constexpr int kB   = 2;
constexpr int kL   = 4096;
constexpr int kD   = 1024;
constexpr int kDqk = 64;
constexpr int kM   = kB * kL;   // 8192 total rows

#if __has_builtin(__builtin_amdgcn_tensor_load_to_lds) && \
    __has_builtin(__builtin_amdgcn_s_wait_tensorcnt)
#define HAVE_TDM 1
#else
#define HAVE_TDM 0
#endif

__device__ inline v8f v8f_zero() {
  v8f z;
#pragma unroll
  for (int i = 0; i < 8; ++i) z[i] = 0.0f;
  return z;
}

// ---- DPP16 16-lane butterfly reductions (no LDS traffic) ----
template <int CTRL>
__device__ inline float dpp_mov(float v) {
#if __has_builtin(__builtin_amdgcn_update_dpp)
  int i = __builtin_bit_cast(int, v);
  int r = __builtin_amdgcn_update_dpp(i, i, CTRL, 0xF, 0xF, true);
  return __builtin_bit_cast(float, r);
#else
  return v;
#endif
}
__device__ inline float wave16_max(float v) {
#if __has_builtin(__builtin_amdgcn_update_dpp)
  v = fmaxf(v, dpp_mov<0xB1>(v));   // quad_perm [1,0,3,2]  : xor 1
  v = fmaxf(v, dpp_mov<0x4E>(v));   // quad_perm [2,3,0,1]  : xor 2
  v = fmaxf(v, dpp_mov<0x141>(v));  // row_half_mirror      : xor 4 (quads uniform)
  v = fmaxf(v, dpp_mov<0x140>(v));  // row_mirror           : xor 8 (octs uniform)
#else
#pragma unroll
  for (int off = 1; off < 16; off <<= 1) v = fmaxf(v, __shfl_xor(v, off, 32));
#endif
  return v;
}
__device__ inline float wave16_sum(float v) {
#if __has_builtin(__builtin_amdgcn_update_dpp)
  v += dpp_mov<0xB1>(v);
  v += dpp_mov<0x4E>(v);
  v += dpp_mov<0x141>(v);
  v += dpp_mov<0x140>(v);
#else
#pragma unroll
  for (int off = 1; off < 16; off <<= 1) v += __shfl_xor(v, off, 32);
#endif
  return v;
}

// ---- Tensor Data Mover: build D# (ISA 08 §8) and issue a 1D/2D bf16 tile load ----
#if HAVE_TDM
__device__ inline void tdm_load_tile(unsigned lds_off, const void* gptr,
                                     unsigned tile0, unsigned tile1,
                                     unsigned long long stride0_elems) {
  unsigned long long ga = (unsigned long long)gptr;
  unsigned dim0 = tile0;
  unsigned dim1 = tile1 ? tile1 : 1u;
  uint32x4 g0;
  g0[0] = 1u;                                         // count=1, user mode
  g0[1] = lds_off;                                    // lds_addr (bytes)
  g0[2] = (unsigned)(ga & 0xFFFFFFFFull);             // global_addr[31:0]
  g0[3] = (unsigned)((ga >> 32) & 0x1FFFFFFull)       // global_addr[56:32]
          | (2u << 30);                               // type = 2 ("image")
  int32x8 g1;
  g1[0] = (int)(1u << 16);                            // data_size = 1 -> 2 bytes
  g1[1] = (int)((dim0 & 0xFFFFu) << 16);              // tensor_dim0[15:0]
  g1[2] = (int)(((dim0 >> 16) & 0xFFFFu) | ((dim1 & 0xFFFFu) << 16));
  g1[3] = (int)(((dim1 >> 16) & 0xFFFFu) | ((tile0 & 0xFFFFu) << 16)); // tile_dim0
  g1[4] = (int)(tile1 & 0xFFFFu);                     // tile_dim1 (0 => 1D), tile_dim2=0
  g1[5] = (int)(unsigned)(stride0_elems & 0xFFFFFFFFull);  // tensor_dim0_stride lo
  g1[6] = (int)(unsigned)((stride0_elems >> 32) & 0xFFFFull);
  g1[7] = 0;
  int32x4 gz = {0, 0, 0, 0};
#if __clang_major__ >= 23
  int32x8 gz8 = {0, 0, 0, 0, 0, 0, 0, 0};
  __builtin_amdgcn_tensor_load_to_lds(g0, g1, gz, gz, gz8, 0);
#else
  __builtin_amdgcn_tensor_load_to_lds(g0, g1, gz, gz, 0);
#endif
}
#endif

// ---------------- conversion kernels ----------------
__global__ void cvt_bf16_kernel(const float* __restrict__ in, bf16* __restrict__ out, int n) {
  int i = blockIdx.x * blockDim.x + threadIdx.x;
  if (i < n) out[i] = (bf16)in[i];
}

// in[rows][cols] (f32) -> out[cols][rows] (bf16): [N,K] B-operand layout
__global__ void transpose_bf16_kernel(const float* __restrict__ in, bf16* __restrict__ out,
                                      int rows, int cols) {
  int i = blockIdx.x * blockDim.x + threadIdx.x;
  if (i < rows * cols) {
    int r = i / cols, c = i % cols;
    out[(size_t)c * rows + r] = (bf16)in[i];
  }
}

// ---------------- generic bf16 WMMA GEMM ----------------
// C = A[M,K] * Bt[N,K]^T.  MODE 0: bf16 out [M,N]; 1: bf16 out [N,M]; 2: f32 [M,N]+bias.
// BM=128, BN template (64/128), BK=32; 256 thr = 8 waves (4 M x 2 N).
// Double-buffered LDS; tiles staged by TDM (wave 0) overlapping compute.
template <int MODE, int BN>
__global__ __launch_bounds__(256) void gemm_kernel(const bf16* __restrict__ A,
                                                   const bf16* __restrict__ Bt,
                                                   bf16* __restrict__ outb,
                                                   float* __restrict__ outf,
                                                   const float* __restrict__ bias,
                                                   int M, int N, int K) {
  constexpr int NF = BN / 32;               // n-frags per wave
  __shared__ bf16 As[2][128 * 32];
  __shared__ bf16 Bs[2][BN * 32];

  const int tid  = threadIdx.x;
  const int lane = tid & 31;
  const int wave = tid >> 5;
  const int wm   = wave >> 1;               // 0..3
  const int wn   = wave & 1;                // 0..1
  const int ln   = lane & 15;
  const int lh   = lane >> 4;

  const int m0 = blockIdx.y * 128;
  const int n0 = blockIdx.x * BN;

  v8f acc[2][NF];
#pragma unroll
  for (int mi = 0; mi < 2; ++mi)
#pragma unroll
    for (int ni = 0; ni < NF; ++ni) acc[mi][ni] = v8f_zero();

  auto stage = [&](int buf, int kk) {
#if HAVE_TDM
    if (wave == 0) {
      tdm_load_tile((unsigned)(size_t)&As[buf][0], A + (size_t)m0 * K + kk,
                    32u, 128u, (unsigned long long)K);
      tdm_load_tile((unsigned)(size_t)&Bs[buf][0], Bt + (size_t)n0 * K + kk,
                    32u, (unsigned)BN, (unsigned long long)K);
    }
#else
    {
      const int arow = tid >> 1, aoff = (tid & 1) * 16;
      const bf16* ap = A + (size_t)(m0 + arow) * K + kk + aoff;
      *(v8bf*)(&As[buf][arow * 32 + aoff])     = *(const v8bf*)(ap);
      *(v8bf*)(&As[buf][arow * 32 + aoff + 8]) = *(const v8bf*)(ap + 8);
      if constexpr (BN == 128) {
        const int brow = tid >> 1, boff = (tid & 1) * 16;
        const bf16* bp = Bt + (size_t)(n0 + brow) * K + kk + boff;
        *(v8bf*)(&Bs[buf][brow * 32 + boff])     = *(const v8bf*)(bp);
        *(v8bf*)(&Bs[buf][brow * 32 + boff + 8]) = *(const v8bf*)(bp + 8);
      } else {
        const int brow = tid >> 2, boff = (tid & 3) * 8;
        const bf16* bp = Bt + (size_t)(n0 + brow) * K + kk + boff;
        *(v8bf*)(&Bs[buf][brow * 32 + boff]) = *(const v8bf*)(bp);
      }
    }
#endif
  };

  const int nk = K / 32;
  stage(0, 0);
  for (int i = 0; i < nk; ++i) {
    if (i + 1 < nk) stage((i + 1) & 1, (i + 1) * 32);
#if HAVE_TDM
    if (wave == 0) {
      if (i + 1 < nk) __builtin_amdgcn_s_wait_tensorcnt(2);  // cur pair done (in-order)
      else            __builtin_amdgcn_s_wait_tensorcnt(0);  // drain
    }
#endif
    __syncthreads();

    const bf16* as = &As[i & 1][0];
    const bf16* bs = &Bs[i & 1][0];
    Frag af[2], bfr[NF];
#pragma unroll
    for (int mi = 0; mi < 2; ++mi) {
      int row = wm * 32 + mi * 16 + ln;
      int c0  = lh * 8;
      af[mi].h[0] = *(const v8bf*)(&as[row * 32 + c0]);
      af[mi].h[1] = *(const v8bf*)(&as[row * 32 + c0 + 16]);
    }
#pragma unroll
    for (int ni = 0; ni < NF; ++ni) {
      int col = wn * (BN / 2) + ni * 16 + ln;
      int kk  = lh * 8;
      bfr[ni].h[0] = *(const v8bf*)(&bs[col * 32 + kk]);
      bfr[ni].h[1] = *(const v8bf*)(&bs[col * 32 + kk + 16]);
    }
#pragma unroll
    for (int mi = 0; mi < 2; ++mi)
#pragma unroll
      for (int ni = 0; ni < NF; ++ni)
        acc[mi][ni] = __builtin_amdgcn_wmma_f32_16x16x32_bf16(
            false, af[mi].v, false, bfr[ni].v, (short)0, acc[mi][ni], false, false);
    __syncthreads();
  }

#pragma unroll
  for (int mi = 0; mi < 2; ++mi) {
#pragma unroll
    for (int ni = 0; ni < NF; ++ni) {
      int gn = n0 + wn * (BN / 2) + ni * 16 + ln;
#pragma unroll
      for (int r = 0; r < 8; ++r) {
        int gm  = m0 + wm * 32 + mi * 16 + lh * 8 + r;
        float v = acc[mi][ni][r];
        if constexpr (MODE == 0)      outb[(size_t)gm * N + gn] = (bf16)v;
        else if constexpr (MODE == 1) outb[(size_t)gn * M + gm] = (bf16)v;
        else                          outf[(size_t)gm * N + gn] = v + bias[gn];
      }
    }
  }
}

// ---------------- fused flash attention ----------------
// grid: (2 column-halves, kM/16 query tiles); block 128 (4 waves).
// 16 query rows, 512 V cols per block; wave w owns 128 cols.
// K tile (32x64) + transposed V tile (512x32) staged in LDS via TDM.
__global__ __launch_bounds__(128) void attn_kernel(const bf16* __restrict__ qb,
                                                   const bf16* __restrict__ kb,
                                                   const bf16* __restrict__ vT,
                                                   bf16* __restrict__ y) {
  __shared__ bf16 Ks[32 * 64];       // [key][dqk]            4 KB
  __shared__ bf16 Vs[512 * 32];      // [local col][key]     32 KB
  __shared__ bf16 Ps[4][16 * 32];    // per-wave P staging    4 KB

  const int tid  = threadIdx.x;
  const int lane = tid & 31;
  const int w    = tid >> 5;
  const int ln   = lane & 15;
  const int lh   = lane >> 4;

  const int ch      = blockIdx.x;
  const int qt      = blockIdx.y;
  const int b       = qt / (kL / 16);
  const int q0      = (qt % (kL / 16)) * 16;
  const int rowbase = b * kL + q0;
  const int colbase = ch * 512 + w * 128;

  // Q fragments for the two 32-wide K steps of DQK=64
  Frag qf[2];
  {
    const bf16* qp = qb + (size_t)(rowbase + ln) * kDqk;
    int c0 = lh * 8;
#pragma unroll
    for (int ks = 0; ks < 2; ++ks) {
      qf[ks].h[0] = *(const v8bf*)(qp + ks * 32 + c0);
      qf[ks].h[1] = *(const v8bf*)(qp + ks * 32 + c0 + 16);
    }
  }

  float m[8], l[8];
  v8f   o[8];
#pragma unroll
  for (int r = 0; r < 8; ++r) { m[r] = -3.0e38f; l[r] = 0.0f; }
#pragma unroll
  for (int c = 0; c < 8; ++c) o[c] = v8f_zero();

  for (int j0 = 0; j0 < kL; j0 += 32) {
    // ---- stage K + V tiles
#if HAVE_TDM
    if (w == 0) {
      tdm_load_tile((unsigned)(size_t)&Ks[0],
                    kb + (size_t)(b * kL + j0) * kDqk, 2048u, 0u, 2048ull);
      tdm_load_tile((unsigned)(size_t)&Vs[0],
                    vT + (size_t)(ch * 512) * kM + b * kL + j0,
                    32u, 512u, (unsigned long long)kM);
      __builtin_amdgcn_s_wait_tensorcnt(0);
    }
#else
    {
      int key = tid >> 2;
      int off = (tid & 3) * 16;
      const bf16* src = kb + (size_t)(b * kL + j0 + key) * kDqk + off;
      *(v8bf*)(&Ks[key * 64 + off])     = *(const v8bf*)(src);
      *(v8bf*)(&Ks[key * 64 + off + 8]) = *(const v8bf*)(src + 8);
      int lc0 = tid * 4;
#pragma unroll
      for (int c = 0; c < 4; ++c) {
        const bf16* vsrc = vT + (size_t)(ch * 512 + lc0 + c) * kM + b * kL + j0;
        bf16* dst = &Vs[(lc0 + c) * 32];
#pragma unroll
        for (int q = 0; q < 4; ++q)
          *(v8bf*)(dst + q * 8) = *(const v8bf*)(vsrc + q * 8);
      }
    }
#endif
    __syncthreads();

    // ---- S = Q K^T  (16 x 32 tile, two 16-wide N chunks)
    v8f s[2];
#pragma unroll
    for (int nc = 0; nc < 2; ++nc) {
      Frag kf0, kf1;
      int coln = nc * 16 + ln;
      int k0   = lh * 8;
      kf0.h[0] = *(const v8bf*)(&Ks[coln * 64 + k0]);
      kf0.h[1] = *(const v8bf*)(&Ks[coln * 64 + k0 + 16]);
      kf1.h[0] = *(const v8bf*)(&Ks[coln * 64 + 32 + k0]);
      kf1.h[1] = *(const v8bf*)(&Ks[coln * 64 + 32 + k0 + 16]);
      v8f acc = v8f_zero();
      acc = __builtin_amdgcn_wmma_f32_16x16x32_bf16(false, qf[0].v, false, kf0.v,
                                                    (short)0, acc, false, false);
      acc = __builtin_amdgcn_wmma_f32_16x16x32_bf16(false, qf[1].v, false, kf1.v,
                                                    (short)0, acc, false, false);
      s[nc] = acc;
    }

    // ---- online softmax (row stats: element r per lane; rows split lo/hi 16 lanes)
    const float scale = 0.125f;  // 1/sqrt(64)
    float alpha[8];
#pragma unroll
    for (int r = 0; r < 8; ++r) {
      float s0   = s[0][r] * scale;
      float s1   = s[1][r] * scale;
      float rm   = wave16_max(fmaxf(s0, s1));
      float mnew = fmaxf(m[r], rm);
      alpha[r]   = __expf(m[r] - mnew);
      float p0   = __expf(s0 - mnew);
      float p1   = __expf(s1 - mnew);
      s[0][r] = p0;
      s[1][r] = p1;
      float rs = wave16_sum(p0 + p1);
      l[r] = l[r] * alpha[r] + rs;
      m[r] = mnew;
    }

    // ---- P: C-layout -> A-fragment layout via per-wave LDS staging
    bf16* ps = &Ps[w][0];
#pragma unroll
    for (int r = 0; r < 8; ++r) {
      int prow = r + lh * 8;
      ps[prow * 32 + ln]      = (bf16)s[0][r];
      ps[prow * 32 + 16 + ln] = (bf16)s[1][r];
    }
    Frag pf;
    {
      int c0 = lh * 8;
      pf.h[0] = *(const v8bf*)(ps + ln * 32 + c0);
      pf.h[1] = *(const v8bf*)(ps + ln * 32 + c0 + 16);
    }

    // ---- O = O*alpha + P @ V
#pragma unroll
    for (int c = 0; c < 8; ++c) {
      Frag vf;
      int col = w * 128 + c * 16 + ln;
      int k0  = lh * 8;
      vf.h[0] = *(const v8bf*)(&Vs[col * 32 + k0]);
      vf.h[1] = *(const v8bf*)(&Vs[col * 32 + k0 + 16]);
      v8f oc = o[c];
#pragma unroll
      for (int r = 0; r < 8; ++r) oc[r] *= alpha[r];
      o[c] = __builtin_amdgcn_wmma_f32_16x16x32_bf16(false, pf.v, false, vf.v,
                                                     (short)0, oc, false, false);
    }
    __syncthreads();
  }

  // ---- normalize and store y (bf16, row-major [kM, kD])
  float inv[8];
#pragma unroll
  for (int r = 0; r < 8; ++r) inv[r] = 1.0f / l[r];
#pragma unroll
  for (int c = 0; c < 8; ++c) {
    int col = colbase + c * 16 + ln;
#pragma unroll
    for (int r = 0; r < 8; ++r) {
      int row = rowbase + r + lh * 8;
      y[(size_t)row * kD + col] = (bf16)(o[c][r] * inv[r]);
    }
  }
}

// ---------------- host orchestration ----------------
extern "C" void kernel_launch(void* const* d_in, const int* in_sizes, int n_in,
                              void* d_out, int out_size, void* d_ws, size_t ws_size,
                              hipStream_t stream) {
  const float* x  = (const float*)d_in[0];
  const float* Wq = (const float*)d_in[1];
  const float* Wk = (const float*)d_in[2];
  const float* Wv = (const float*)d_in[3];
  const float* Wo = (const float*)d_in[4];
  const float* bo = (const float*)d_in[5];
  float* out = (float*)d_out;

  char* p = (char*)d_ws;
  auto carve = [&](size_t elems) -> bf16* {
    bf16* r = (bf16*)p;
    p += (elems * sizeof(bf16) + 255) & ~(size_t)255;
    return r;
  };
  bf16* xb   = carve((size_t)kM * kD);
  bf16* WqT  = carve((size_t)kDqk * kD);
  bf16* WkT  = carve((size_t)kDqk * kD);
  bf16* WvT  = carve((size_t)kD * kD);
  bf16* WoT  = carve((size_t)kD * kD);
  bf16* qbuf = carve((size_t)kM * kDqk);
  bf16* kbuf = carve((size_t)kM * kDqk);
  bf16* vT   = carve((size_t)kD * kM);
  bf16* yb   = carve((size_t)kM * kD);

  {
    int n = kM * kD;
    cvt_bf16_kernel<<<(n + 255) / 256, 256, 0, stream>>>(x, xb, n);
    transpose_bf16_kernel<<<(kD * kDqk + 255) / 256, 256, 0, stream>>>(Wq, WqT, kD, kDqk);
    transpose_bf16_kernel<<<(kD * kDqk + 255) / 256, 256, 0, stream>>>(Wk, WkT, kD, kDqk);
    transpose_bf16_kernel<<<(kD * kD + 255) / 256, 256, 0, stream>>>(Wv, WvT, kD, kD);
    transpose_bf16_kernel<<<(kD * kD + 255) / 256, 256, 0, stream>>>(Wo, WoT, kD, kD);
  }

  dim3 blk(256);
  gemm_kernel<0, 64><<<dim3(kDqk / 64, kM / 128), blk, 0, stream>>>(
      xb, WqT, qbuf, nullptr, nullptr, kM, kDqk, kD);
  gemm_kernel<0, 64><<<dim3(kDqk / 64, kM / 128), blk, 0, stream>>>(
      xb, WkT, kbuf, nullptr, nullptr, kM, kDqk, kD);
  gemm_kernel<1, 128><<<dim3(kD / 128, kM / 128), blk, 0, stream>>>(
      xb, WvT, vT, nullptr, nullptr, kM, kD, kD);
  attn_kernel<<<dim3(2, kM / 16), dim3(128), 0, stream>>>(qbuf, kbuf, vT, yb);
  gemm_kernel<2, 128><<<dim3(kD / 128, kM / 128), blk, 0, stream>>>(
      yb, WoT, nullptr, out, bo, kM, kD, kD);
}
